// WFR_Module_64639257805136
// MI455X (gfx1250) — compile-verified, matrix operand order checked by reading, not run on hardware
//
#include <hip/hip_runtime.h>
#include <hip/hip_bf16.h>

// Select WMMA flavor per compilation pass. The host pass must never evaluate
// device builtins via #error; it just needs to parse one valid path.
#if defined(__HIP_DEVICE_COMPILE__) && __has_builtin(__builtin_amdgcn_wmma_f32_16x16x4_f32)
#define USE_WMMA_F32 1
#else
#define USE_WMMA_F32 0
#endif

typedef float    v2f  __attribute__((ext_vector_type(2)));
typedef float    v8f  __attribute__((ext_vector_type(8)));
typedef _Float16 v16h __attribute__((ext_vector_type(16)));

// problem dims (fixed by setup_inputs)
constexpr int NB  = 8;
constexpr int NC  = 192;
constexpr int NH  = 128;
constexpr int NW  = 128;
constexpr int HH  = 64;      // h
constexpr int WW  = 64;      // w
constexpr int NT  = 4096;    // h*w
constexpr int NBT = NB * NT; // 32768
constexpr int NC3 = 576;
constexpr int NHC = 768;
constexpr int NHW = NH * NW; // 16384
constexpr int NCHK = 32;     // wkv chunks
constexpr int CHL  = 128;    // chunk length

// GEMM epilogue ops
#define EP_STORE    0
#define EP_SIGMOID  1
#define EP_RELUSQ   2
#define EP_ADDTO    3   // Out += acc
#define EP_MULADD   4   // Out += aux * acc
#define EP_MULSTORE 5   // Out  = (acc+bias) * aux

// ---------------------------------------------------------------------------
// Haar forward: x(NCHW) -> s0[BT][C] (LL, via LDS transpose) + xh(B,3C,h,w)
// grid: (NB*HH, 12), block (32,32): blockIdx.y = ct*2 + jt
// ---------------------------------------------------------------------------
__global__ __launch_bounds__(1024) void haar_fwd(const float* __restrict__ x,
                                                 float* __restrict__ s0,
                                                 float* __restrict__ xh) {
  __shared__ float tl[32][33];
  const int tx = threadIdx.x, ty = threadIdx.y;
  const int b  = blockIdx.x / HH;
  const int i  = blockIdx.x % HH;
  const int ct = blockIdx.y >> 1;
  const int jt = blockIdx.y & 1;
  const int c  = ct * 32 + ty;
  const int j  = jt * 32 + tx;

  const float* xp = x + (((size_t)b * NC + c) * NH + 2 * i) * NW + 2 * j;
  const float v00 = xp[0], v01 = xp[1], v10 = xp[NW], v11 = xp[NW + 1];
  const float ll = 0.5f * (v00 + v01 + v10 + v11);
  const float s1 = 0.5f * (v00 + v01 - v10 - v11);  // hi(row) x lo(col)
  const float s2 = 0.5f * (v00 - v01 + v10 - v11);  // lo x hi
  const float s3 = 0.5f * (v00 - v01 - v10 + v11);  // hi x hi

  const size_t hb = (((size_t)b * NC3 + (size_t)c * 3) * HH + i) * WW + j;
  xh[hb]                       = s1;
  xh[hb + (size_t)HH * WW]     = s2;
  xh[hb + (size_t)2 * HH * WW] = s3;

  tl[ty][tx] = ll;
  __syncthreads();
  // transposed write: token-major
  s0[((size_t)b * NT + (size_t)i * WW + jt * 32 + ty) * NC + ct * 32 + tx] = tl[tx][ty];
}

// ---------------------------------------------------------------------------
// LayerNorm over C per row; optional elementwise multiplier (for sr * ln(y))
// 8 rows / block (one wave32 per row)
// ---------------------------------------------------------------------------
__global__ __launch_bounds__(256) void ln_rows(const float* __restrict__ in,
                                               float* __restrict__ out,
                                               const float* __restrict__ g,
                                               const float* __restrict__ bb,
                                               const float* __restrict__ mulp,
                                               int rows) {
  const int wave = threadIdx.x >> 5, lane = threadIdx.x & 31;
  const int row  = blockIdx.x * 8 + wave;
  if (row >= rows) return;
  const float* rp = in + (size_t)row * NC;
  float vals[6];
  float sum = 0.f;
  int nv = 0;
  for (int c = lane; c < NC; c += 32) { vals[nv] = rp[c]; sum += vals[nv]; ++nv; }
#pragma unroll
  for (int off = 16; off; off >>= 1) sum += __shfl_xor(sum, off, 32);
  const float mean = sum * (1.f / NC);
  float var = 0.f;
#pragma unroll
  for (int i = 0; i < 6; ++i) { float d = vals[i] - mean; var += d * d; }
#pragma unroll
  for (int off = 16; off; off >>= 1) var += __shfl_xor(var, off, 32);
  const float inv = rsqrtf(var * (1.f / NC) + 1e-5f);
  float* op = out + (size_t)row * NC;
  nv = 0;
  for (int c = lane; c < NC; c += 32) {
    float v = (vals[nv] - mean) * inv * g[c] + bb[c];
    if (mulp) v *= mulp[(size_t)row * NC + c];
    op[c] = v;
    ++nv;
  }
}

__global__ __launch_bounds__(256) void zero_k(float* p, int n) {
  int i = blockIdx.x * 256 + threadIdx.x;
  if (i < n) p[i] = 0.f;
}

// gc[b][c] = sum_t s[b*T+t][c] ; grid NB*16 blocks of 192 threads
__global__ __launch_bounds__(192) void gc_reduce(const float* __restrict__ s,
                                                 float* __restrict__ gc) {
  const int b = blockIdx.x >> 4;
  const int chunk = blockIdx.x & 15;
  const int c = threadIdx.x;
  float acc = 0.f;
  const int t0 = chunk * (NT / 16);
  for (int t = t0; t < t0 + NT / 16; ++t)
    acc += s[((size_t)b * NT + t) * NC + c];
  atomicAdd(&gc[b * NC + c], acc);
}

// decay-gate MLP + wdec; single block
__global__ __launch_bounds__(256) void decay_gate(const float* __restrict__ gc,
                                                  const float* __restrict__ w1,
                                                  const float* __restrict__ b1,
                                                  const float* __restrict__ w2,
                                                  const float* __restrict__ b2,
                                                  const float* __restrict__ smd,
                                                  float* __restrict__ wdec) {
  __shared__ float gcm[NB * NC];
  __shared__ float hid[NB * 48];
  const int tid = threadIdx.x;
  for (int i = tid; i < NB * NC; i += 256) gcm[i] = gc[i] * (1.f / NT);
  __syncthreads();
  for (int i = tid; i < NB * 48; i += 256) {
    const int b = i / 48, j = i % 48;
    float a = b1[j];
    for (int c = 0; c < NC; ++c) a += gcm[b * NC + c] * w1[j * NC + c];
    hid[i] = tanhf(a);
  }
  __syncthreads();
  if (tid < NC) {
    float acc = 0.f;
    for (int b = 0; b < NB; ++b) {
      float d = b2[tid];
      for (int j = 0; j < 48; ++j) d += hid[b * 48 + j] * w2[tid * 48 + j];
      acc += 1.f / (1.f + __expf(-d));
    }
    wdec[tid] = smd[tid] + acc * (1.f / NB);
  }
}

// q-shift + token mixes -> xk, xv, xr
__global__ __launch_bounds__(256) void qshift_mix(const float* __restrict__ s,
                                                  const float* __restrict__ smk,
                                                  const float* __restrict__ smv,
                                                  const float* __restrict__ smr,
                                                  float* __restrict__ xk,
                                                  float* __restrict__ xv,
                                                  float* __restrict__ xr,
                                                  int total) {
  const int idx = blockIdx.x * 256 + threadIdx.x;
  if (idx >= total) return;
  const int c  = idx % NC;
  const int bt = idx / NC;
  const int t  = bt & (NT - 1);
  const int b  = bt >> 12;
  const int i  = t >> 6, j = t & 63;
  const int g  = c / 48;
  int ti = i, tj = j;
  bool ok;
  if (g == 0)      { tj = j - 1; ok = (j > 0); }
  else if (g == 1) { tj = j + 1; ok = (j < WW - 1); }
  else if (g == 2) { ti = i - 1; ok = (i > 0); }
  else             { ti = i + 1; ok = (i < HH - 1); }
  float xx = 0.f;
  if (ok) xx = s[(((size_t)b << 12) + (ti << 6) + tj) * NC + c];
  const float sv = s[idx];
  const float mk = smk[c], mv = smv[c], mr = smr[c];
  xk[idx] = sv * mk + xx * (1.f - mk);
  xv[idx] = sv * mv + xx * (1.f - mv);
  xr[idx] = sv * mr + xx * (1.f - mr);
}

// ---------------------------------------------------------------------------
// WKV chunked scan (log-sum-exp stable, same numerics structure as reference)
// ---------------------------------------------------------------------------
__global__ __launch_bounds__(192) void wkv_chunk(const float* __restrict__ k,
                                                 const float* __restrict__ v,
                                                 const float* __restrict__ wdec,
                                                 float* __restrict__ ca,
                                                 float* __restrict__ cb,
                                                 float* __restrict__ cp) {
  const int b = blockIdx.x / NCHK;
  const int ch = blockIdx.x % NCHK;
  const int c = threadIdx.x;
  const float w = wdec[c];
  float aa = 0.f, bb = 0.f, pp = -1e38f;
  size_t base = ((size_t)b * NT + (size_t)ch * CHL) * NC + c;
#pragma unroll 4
  for (int t = 0; t < CHL; ++t) {
    const float kt = k[base], vt = v[base];
    const float ww = w + pp;
    const float q  = fmaxf(ww, kt);
    const float e1 = __expf(ww - q), e2 = __expf(kt - q);
    aa = e1 * aa + e2 * vt;
    bb = e1 * bb + e2;
    pp = q;
    base += NC;
  }
  const size_t o = ((size_t)b * NCHK + ch) * NC + c;
  ca[o] = aa; cb[o] = bb; cp[o] = pp;
}

__global__ __launch_bounds__(256) void wkv_combine(const float* __restrict__ ca,
                                                   const float* __restrict__ cb,
                                                   const float* __restrict__ cp,
                                                   float* __restrict__ ra,
                                                   float* __restrict__ rb,
                                                   float* __restrict__ rp,
                                                   const float* __restrict__ wdec) {
  const int idx = blockIdx.x * 256 + threadIdx.x;
  if (idx >= NB * NC) return;
  const int c = idx % NC;
  const int b = idx / NC;
  const float wL = wdec[c] * (float)CHL;
  float aa = 0.f, bb = 0.f, pp = -1e38f;
  for (int n = 0; n < NCHK; ++n) {
    const size_t o = ((size_t)b * NCHK + n) * NC + c;
    ra[o] = aa; rb[o] = bb; rp[o] = pp;      // carry-in for chunk n
    const float sh = wL + pp;
    const float ps = cp[o];
    const float q  = fmaxf(sh, ps);
    const float e1 = __expf(sh - q), e2 = __expf(ps - q);
    aa = e1 * aa + e2 * ca[o];
    bb = e1 * bb + e2 * cb[o];
    pp = q;
  }
}

__global__ __launch_bounds__(192) void wkv_out(const float* __restrict__ k,
                                               const float* __restrict__ v,
                                               const float* __restrict__ wdec,
                                               const float* __restrict__ u,
                                               const float* __restrict__ ra,
                                               const float* __restrict__ rb,
                                               const float* __restrict__ rp,
                                               float* __restrict__ y) {
  const int b = blockIdx.x / NCHK;
  const int ch = blockIdx.x % NCHK;
  const int c = threadIdx.x;
  const float w = wdec[c], uu = u[c];
  const size_t co = ((size_t)b * NCHK + ch) * NC + c;
  float aa = ra[co], bb = rb[co], pp = rp[co];
  size_t base = ((size_t)b * NT + (size_t)ch * CHL) * NC + c;
#pragma unroll 2
  for (int t = 0; t < CHL; ++t) {
    const float kt = k[base], vt = v[base];
    float ww = uu + kt;
    float q  = fmaxf(pp, ww);
    float e1 = __expf(pp - q), e2 = __expf(ww - q);
    y[base] = (e1 * aa + e2 * vt) / (e1 * bb + e2);
    ww = w + pp;
    q  = fmaxf(ww, kt);
    e1 = __expf(ww - q); e2 = __expf(kt - q);
    aa = e1 * aa + e2 * vt;
    bb = e1 * bb + e2;
    pp = q;
    base += NC;
  }
}

// ---------------------------------------------------------------------------
// Generic WMMA GEMM:  Out[M][N] (+=|=) epilogue( A[M][K] @ W[N][K]^T )
// block tile 128x64, 8 waves, wave tile 32x32, K-tile 32.
// Device path A: V_WMMA_F32_16X16X4_F32 (full f32 precision) if available.
// Device path B: V_WMMA_F32_16X16X32_F16 (f16 in, f32 accumulate) otherwise.
// ---------------------------------------------------------------------------
__device__ __forceinline__ v16h pack_frag_f16(const float* rowp, int lh) {
  // 16-bit A/B fragment layout (16x32): lanes 0-15 hold K 0..7 / 16..23,
  // lanes 16-31 hold K 8..15 / 24..31 (lh = lane>>4 selects the K-half).
  v16h f;
#pragma unroll
  for (int e = 0; e < 8; ++e) {
    f[e]     = (_Float16)rowp[lh * 8 + e];
    f[e + 8] = (_Float16)rowp[16 + lh * 8 + e];
  }
  return f;
}

__global__ __launch_bounds__(256) void gemm_f32_wmma(const float* __restrict__ A,
                                                     const float* __restrict__ Wt,
                                                     float* __restrict__ Out,
                                                     const float* __restrict__ bias,
                                                     const float* __restrict__ aux,
                                                     int M, int N, int K, int op) {
  __shared__ float lA[128][33];
  __shared__ float lW[64][33];
  const int tid  = threadIdx.x;
  const int lane = tid & 31, wave = tid >> 5;
  const int wm = wave >> 1, wn = wave & 1;
  const int bm = blockIdx.x * 128, bn = blockIdx.y * 64;
  const int l16 = lane & 15;
  const int lh  = lane >> 4;

  v8f acc[2][2];
#pragma unroll
  for (int a = 0; a < 2; ++a)
#pragma unroll
    for (int b = 0; b < 2; ++b)
#pragma unroll
      for (int r = 0; r < 8; ++r) acc[a][b][r] = 0.f;

  for (int kk = 0; kk < K; kk += 32) {
    // A tile 128x32 = 1024 float4, 4 per thread
#pragma unroll
    for (int r = 0; r < 4; ++r) {
      const int idx = tid * 4 + r;
      const int row = idx >> 3, c4 = (idx & 7) << 2;
      const float4 v4 = *(const float4*)(A + (size_t)(bm + row) * K + kk + c4);
      lA[row][c4 + 0] = v4.x; lA[row][c4 + 1] = v4.y;
      lA[row][c4 + 2] = v4.z; lA[row][c4 + 3] = v4.w;
    }
    // W tile 64x32 = 512 float4, 2 per thread
#pragma unroll
    for (int r = 0; r < 2; ++r) {
      const int idx = tid * 2 + r;
      const int row = idx >> 3, c4 = (idx & 7) << 2;
      const float4 v4 = *(const float4*)(Wt + (size_t)(bn + row) * K + kk + c4);
      lW[row][c4 + 0] = v4.x; lW[row][c4 + 1] = v4.y;
      lW[row][c4 + 2] = v4.z; lW[row][c4 + 3] = v4.w;
    }
    if (kk + 32 < K)  // prefetch next A tile (global_prefetch_b8)
      __builtin_prefetch(A + (size_t)(bm + (tid >> 1)) * K + kk + 32, 0, 3);
    __syncthreads();

#if USE_WMMA_F32
    const int koff = lh << 1;
#pragma unroll
    for (int k4 = 0; k4 < 32; k4 += 4) {
      v2f a0, a1, b0, b1;
      a0[0] = lA[wm * 32 + l16][k4 + koff];
      a0[1] = lA[wm * 32 + l16][k4 + koff + 1];
      a1[0] = lA[wm * 32 + 16 + l16][k4 + koff];
      a1[1] = lA[wm * 32 + 16 + l16][k4 + koff + 1];
      b0[0] = lW[wn * 32 + l16][k4 + koff];
      b0[1] = lW[wn * 32 + l16][k4 + koff + 1];
      b1[0] = lW[wn * 32 + 16 + l16][k4 + koff];
      b1[1] = lW[wn * 32 + 16 + l16][k4 + koff + 1];
      acc[0][0] = __builtin_amdgcn_wmma_f32_16x16x4_f32(false, a0, false, b0, (short)0, acc[0][0], false, false);
      acc[0][1] = __builtin_amdgcn_wmma_f32_16x16x4_f32(false, a0, false, b1, (short)0, acc[0][1], false, false);
      acc[1][0] = __builtin_amdgcn_wmma_f32_16x16x4_f32(false, a1, false, b0, (short)0, acc[1][0], false, false);
      acc[1][1] = __builtin_amdgcn_wmma_f32_16x16x4_f32(false, a1, false, b1, (short)0, acc[1][1], false, false);
    }
#else
    {
      const v16h a0 = pack_frag_f16(&lA[wm * 32 + l16][0], lh);
      const v16h a1 = pack_frag_f16(&lA[wm * 32 + 16 + l16][0], lh);
      const v16h b0 = pack_frag_f16(&lW[wn * 32 + l16][0], lh);
      const v16h b1 = pack_frag_f16(&lW[wn * 32 + 16 + l16][0], lh);
      acc[0][0] = __builtin_amdgcn_wmma_f32_16x16x32_f16(false, a0, false, b0, (short)0, acc[0][0], false, false);
      acc[0][1] = __builtin_amdgcn_wmma_f32_16x16x32_f16(false, a0, false, b1, (short)0, acc[0][1], false, false);
      acc[1][0] = __builtin_amdgcn_wmma_f32_16x16x32_f16(false, a1, false, b0, (short)0, acc[1][0], false, false);
      acc[1][1] = __builtin_amdgcn_wmma_f32_16x16x32_f16(false, a1, false, b1, (short)0, acc[1][1], false, false);
    }
#endif
    __syncthreads();
  }

#pragma unroll
  for (int mi = 0; mi < 2; ++mi)
#pragma unroll
    for (int ni = 0; ni < 2; ++ni) {
      const int n = bn + wn * 32 + ni * 16 + l16;
      const float bv = bias ? bias[n] : 0.f;
      const int mbase = bm + wm * 32 + mi * 16 + lh * 8;
#pragma unroll
      for (int r = 0; r < 8; ++r) {
        const size_t o = (size_t)(mbase + r) * N + n;
        const float v = acc[mi][ni][r] + bv;
        if (op == EP_STORE)        Out[o] = v;
        else if (op == EP_SIGMOID) Out[o] = 1.f / (1.f + __expf(-v));
        else if (op == EP_RELUSQ)  { const float t = v > 0.f ? v : 0.f; Out[o] = t * t; }
        else if (op == EP_ADDTO)   Out[o] = Out[o] + v;
        else if (op == EP_MULADD)  Out[o] = Out[o] + aux[o] * v;
        else /* EP_MULSTORE */     Out[o] = v * aux[o];
      }
    }
}

// ---------------------------------------------------------------------------
// Grouped 3x3 conv (groups=C, 3 in / 3 out per group) + BN + LeakyReLU(0.01)
// xh NCHW -> hd NHWC [BT][576]; grid (NB*HH, 12): blockIdx.y = ct*2 + jt
// ---------------------------------------------------------------------------
__global__ __launch_bounds__(256) void dw_conv(const float* __restrict__ xh,
                                               const float* __restrict__ dw,
                                               const float* __restrict__ bg,
                                               const float* __restrict__ bbb,
                                               const float* __restrict__ bm,
                                               const float* __restrict__ bv,
                                               float* __restrict__ hd) {
  __shared__ float patch[96][3][34];
  __shared__ float outs[32][96];
  const int tid = threadIdx.x;
  const int b  = blockIdx.x / HH;
  const int i  = blockIdx.x % HH;
  const int ct = blockIdx.y >> 1;
  const int jt = blockIdx.y & 1;
  const int ch0 = ct * 96;
  const int j0  = jt * 32;

  for (int idx = tid; idx < 96 * 3 * 34; idx += 256) {
    const int cl = idx / 102;
    const int rem = idx % 102;
    const int ky = rem / 34;
    const int xx = rem % 34;
    const int gi = i + ky - 1;
    const int gj = j0 + xx - 1;
    float val = 0.f;
    if (gi >= 0 && gi < HH && gj >= 0 && gj < WW)
      val = xh[(((size_t)b * NC3 + ch0 + cl) * HH + gi) * WW + gj];
    patch[cl][ky][xx] = val;
  }
  __syncthreads();

  const int jl = tid & 31;
  const int cw = tid >> 5;  // 0..7
  for (int co = cw; co < 96; co += 8) {
    const int gbase = (co / 3) * 3;  // local input-channel base of the group
    const float* wt = dw + (size_t)(ch0 + co) * 27;
    float acc = 0.f;
#pragma unroll
    for (int i2 = 0; i2 < 3; ++i2)
#pragma unroll
      for (int ky = 0; ky < 3; ++ky)
#pragma unroll
        for (int kx = 0; kx < 3; ++kx)
          acc += wt[(i2 * 3 + ky) * 3 + kx] * patch[gbase + i2][ky][jl + kx];
    const int ch = ch0 + co;
    float o = (acc - bm[ch]) * rsqrtf(bv[ch] + 1e-5f) * bg[ch] + bbb[ch];
    o = (o >= 0.f) ? o : 0.01f * o;
    outs[jl][co] = o;
  }
  __syncthreads();
  for (int idx = tid; idx < 32 * 96; idx += 256) {
    const int jj = idx / 96, co = idx % 96;
    hd[((size_t)b * NT + (size_t)i * WW + j0 + jj) * NC3 + ch0 + co] = outs[jj][co];
  }
}

// Orthonormal inverse Haar: s (LL) + gated (3 bands, NHWC, ch=c*3+k) -> NHWC full-res
__global__ __launch_bounds__(256) void inv_haar(const float* __restrict__ s,
                                                const float* __restrict__ gated,
                                                float* __restrict__ outp,
                                                int total) {
  const int idx = blockIdx.x * 256 + threadIdx.x;
  if (idx >= total) return;
  const int c  = idx % NC;
  const int bt = idx / NC;
  const int t  = bt & (NT - 1);
  const int b  = bt >> 12;
  const int i  = t >> 6, j = t & 63;
  const float LL = s[idx];
  const float* gp = gated + (size_t)bt * NC3 + (size_t)c * 3;
  const float S1 = gp[0], S2 = gp[1], S3 = gp[2];
  const float o00 = 0.5f * (LL + S1 + S2 + S3);
  const float o01 = 0.5f * (LL + S1 - S2 - S3);
  const float o10 = 0.5f * (LL - S1 + S2 - S3);
  const float o11 = 0.5f * (LL - S1 - S2 + S3);
  float* op = outp + (((size_t)b * NHW) + (size_t)(2 * i) * NW + 2 * j) * NC + c;
  op[0]                    = o00;
  op[NC]                   = o01;
  op[(size_t)NW * NC]      = o10;
  op[(size_t)NW * NC + NC] = o11;
}

// NHWC -> NCHW transpose with residual add; block (32,32)
__global__ __launch_bounds__(1024) void transpose_add(const float* __restrict__ src,
                                                      const float* __restrict__ x,
                                                      float* __restrict__ out) {
  __shared__ float tl[32][33];
  const int tx = threadIdx.x, ty = threadIdx.y;
  const int b  = blockIdx.x;
  const int p0 = blockIdx.y * 32;
  const int c0 = blockIdx.z * 32;
  tl[ty][tx] = src[((size_t)b * NHW + p0 + ty) * NC + c0 + tx];
  __syncthreads();
  const int c = c0 + ty;
  const int p = p0 + tx;
  const size_t o = ((size_t)b * NC + c) * NHW + p;
  out[o] = tl[tx][ty] + x[o];
}

// ---------------------------------------------------------------------------
extern "C" void kernel_launch(void* const* d_in, const int* in_sizes, int n_in,
                              void* d_out, int out_size, void* d_ws, size_t ws_size,
                              hipStream_t stream) {
  (void)in_sizes; (void)n_in; (void)out_size; (void)ws_size;
  const float* x        = (const float*)d_in[0];
  const float* ln1_g    = (const float*)d_in[1];
  const float* ln1_b    = (const float*)d_in[2];
  const float* ln2_g    = (const float*)d_in[3];
  const float* ln2_b    = (const float*)d_in[4];
  const float* sm_decay = (const float*)d_in[5];
  const float* sm_first = (const float*)d_in[6];
  const float* smk      = (const float*)d_in[7];
  const float* smv      = (const float*)d_in[8];
  const float* smr      = (const float*)d_in[9];
  const float* wk       = (const float*)d_in[10];
  const float* wv       = (const float*)d_in[11];
  const float* wr       = (const float*)d_in[12];
  const float* wo       = (const float*)d_in[13];
  const float* kn_g     = (const float*)d_in[14];
  const float* kn_b     = (const float*)d_in[15];
  const float* dg_w1    = (const float*)d_in[16];
  const float* dg_b1    = (const float*)d_in[17];
  const float* dg_w2    = (const float*)d_in[18];
  const float* dg_b2    = (const float*)d_in[19];
  // d_in[20] cmk, d_in[21] cmr: algebraically identity (s*m + s*(1-m) == s)
  const float* cm_key   = (const float*)d_in[22];
  const float* cm_rec   = (const float*)d_in[23];
  const float* cm_val   = (const float*)d_in[24];
  const float* dw_w     = (const float*)d_in[25];
  const float* bn_g     = (const float*)d_in[26];
  const float* bn_b     = (const float*)d_in[27];
  const float* bn_m     = (const float*)d_in[28];
  const float* bn_v     = (const float*)d_in[29];
  const float* pw_w     = (const float*)d_in[30];
  const float* pw_b     = (const float*)d_in[31];
  const float* gate_w   = (const float*)d_in[32];
  const float* gate_b   = (const float*)d_in[33];
  const float* fus_w    = (const float*)d_in[34];
  const float* fus_b    = (const float*)d_in[35];

  float* ws = (float*)d_ws;
  const size_t SEQ = (size_t)NBT * NC;   // 6,291,456 floats
  const size_t BIG = (size_t)NBT * NC3;  // 18,874,368 floats
  float* s    = ws;
  float* xk   = ws + 1 * SEQ;
  float* xv   = ws + 2 * SEQ;
  float* xr   = ws + 3 * SEQ;
  float* kb   = ws + 4 * SEQ;
  float* vb   = ws + 5 * SEQ;
  float* rb   = ws + 6 * SEQ;
  float* yb   = ws + 7 * SEQ;
  float* xh   = ws + 8 * SEQ;
  float* hd   = xh + BIG;
  float* gate = hd + BIG;
  float* gated= gate + BIG;
  float* sm   = gated + BIG;
  float* gc   = sm;                 // NB*NC
  float* wdec = gc + NB * NC;       // NC
  const size_t SC = (size_t)NB * NCHK * NC;
  float* csa = wdec + NC;
  float* csb = csa + SC;
  float* csp = csb + SC;
  float* cra = csp + SC;
  float* crb = cra + SC;
  float* crp = crb + SC;
  // stage-reused aliases
  float* k2   = xk;           // spans xk..kb (BT x 768), live only in channel-mix
  float* r2   = vb;           // v dead after wkv_out
  float* outp = xk;           // spans xk..kb (B*HW x C), live only at the end
  float* fuso = vb;           // spans vb..yb + first SEQ of xh (all dead then)

  const dim3 b32(32, 32);
  const int tot = NBT * NC;

  haar_fwd<<<dim3(NB * HH, 12), b32, 0, stream>>>(x, s, xh);
  ln_rows<<<NBT / 8, 256, 0, stream>>>(s, s, ln1_g, ln1_b, nullptr, NBT);
  zero_k<<<(NB * NC + 255) / 256, 256, 0, stream>>>(gc, NB * NC);
  gc_reduce<<<NB * 16, 192, 0, stream>>>(s, gc);
  decay_gate<<<1, 256, 0, stream>>>(gc, dg_w1, dg_b1, dg_w2, dg_b2, sm_decay, wdec);
  qshift_mix<<<(tot + 255) / 256, 256, 0, stream>>>(s, smk, smv, smr, xk, xv, xr, tot);

  const dim3 gCC(NBT / 128, NC / 64);
  gemm_f32_wmma<<<gCC, 256, 0, stream>>>(xk, wk, kb, nullptr, nullptr, NBT, NC, NC, EP_STORE);
  gemm_f32_wmma<<<gCC, 256, 0, stream>>>(xv, wv, vb, nullptr, nullptr, NBT, NC, NC, EP_STORE);
  gemm_f32_wmma<<<gCC, 256, 0, stream>>>(xr, wr, rb, nullptr, nullptr, NBT, NC, NC, EP_SIGMOID);

  wkv_chunk<<<NB * NCHK, 192, 0, stream>>>(kb, vb, wdec, csa, csb, csp);
  wkv_combine<<<(NB * NC + 255) / 256, 256, 0, stream>>>(csa, csb, csp, cra, crb, crp, wdec);
  wkv_out<<<NB * NCHK, 192, 0, stream>>>(kb, vb, wdec, sm_first, cra, crb, crp, yb);

  // ym = sr * LN_kn(y)  (python precedence: (sr*y) @ wo.T)
  ln_rows<<<NBT / 8, 256, 0, stream>>>(yb, yb, kn_g, kn_b, rb, NBT);
  gemm_f32_wmma<<<gCC, 256, 0, stream>>>(yb, wo, s, nullptr, nullptr, NBT, NC, NC, EP_ADDTO);
  ln_rows<<<NBT / 8, 256, 0, stream>>>(s, s, ln2_g, ln2_b, nullptr, NBT);

  // channel mix (cmk/cmr cancel)
  gemm_f32_wmma<<<dim3(NBT / 128, NHC / 64), 256, 0, stream>>>(s, cm_key, k2, nullptr, nullptr, NBT, NHC, NC, EP_RELUSQ);
  gemm_f32_wmma<<<gCC, 256, 0, stream>>>(s, cm_rec, r2, nullptr, nullptr, NBT, NC, NC, EP_SIGMOID);
  gemm_f32_wmma<<<gCC, 256, 0, stream>>>(k2, cm_val, s, nullptr, r2, NBT, NC, NHC, EP_MULADD);

  // high path
  gemm_f32_wmma<<<dim3(NBT / 128, NC3 / 64), 256, 0, stream>>>(s, gate_w, gate, gate_b, nullptr, NBT, NC3, NC, EP_SIGMOID);
  dw_conv<<<dim3(NB * HH, 12), 256, 0, stream>>>(xh, dw_w, bn_g, bn_b, bn_m, bn_v, hd);
  gemm_f32_wmma<<<dim3(NBT / 128, NC3 / 64), 256, 0, stream>>>(hd, pw_w, gated, pw_b, gate, NBT, NC3, NC3, EP_MULSTORE);

  // reconstruction + fusion + residual
  inv_haar<<<(tot + 255) / 256, 256, 0, stream>>>(s, gated, outp, tot);
  gemm_f32_wmma<<<dim3((NB * NHW) / 128, NC / 64), 256, 0, stream>>>(outp, fus_w, fuso, fus_b, nullptr, NB * NHW, NC, NC, EP_STORE);
  transpose_add<<<dim3(NB, NHW / 32, NC / 32), b32, 0, stream>>>(fuso, x, (float*)d_out);
}
